// SynapticChain_47648367182491
// MI455X (gfx1250) — matrix operand order repaired and checked by reading, not compiled, not run on hardware
//
#include <hip/hip_runtime.h>

typedef __attribute__((ext_vector_type(16))) _Float16 v16h;
typedef __attribute__((ext_vector_type(8)))  float    v8f;

#define B_      4
#define T_      24
#define N_      4096
#define F_      32
#define LAYERS  3
#define WAVES   8
#define TILES_PER_B (N_ / 16)   // 256 row-tiles per batch element

__device__ __forceinline__ v8f wmma32(v16h a, v16h b, v8f c) {
  // D = A(16x32 f16) * B(32x16 f16) + C(16x16 f32)
  return __builtin_amdgcn_wmma_f32_16x16x32_f16(
      /*neg_a=*/false, a, /*neg_b=*/false, b,
      /*c_mod=*/(short)0, c, /*reuse_a=*/false, /*reuse_b=*/false);
}

// WMMA with inline-0 accumulator (no C registers: SRC2 encodes inline 0)
__device__ __forceinline__ v8f wmma32_z(v16h a, v16h b) {
  v8f z = {};
  return __builtin_amdgcn_wmma_f32_16x16x32_f16(
      false, a, false, b, (short)0, z, false, false);
}

__device__ __forceinline__ float clamp01(float v) {
  return fminf(fmaxf(v, 0.0f), 1.0f);
}

// Load a 16-bit A fragment (16 x 32 slice) from an LDS row-major f16 buffer
// with row stride 64 elements.  ISA 16-bit A layout (wave32):
//   lane l: row m = l&15, half hh = l>>4
//   vgpr j (halves 2j,2j+1): K = 8*hh + 2j          (j<4)
//                            K = 16 + 8*hh + 2(j-4) (j>=4)
// Pairs of adjacent K are dword-aligned -> compiler emits 2x ds_load_b128.
__device__ __forceinline__ v16h loadA(const _Float16* sb, int m, int hh, int koff) {
  v16h a;
#pragma unroll
  for (int j = 0; j < 8; ++j) {
    int k = koff + ((j < 4) ? (8 * hh + 2 * j) : (16 + 8 * hh + 2 * (j - 4)));
    a[2 * j]     = sb[m * 64 + k];
    a[2 * j + 1] = sb[m * 64 + k + 1];
  }
  return a;
}

__global__ __launch_bounds__(256) void snn_chain_kernel(
    const float* __restrict__ x,
    const float* __restrict__ alphas,
    const float* __restrict__ betas,
    const float* __restrict__ thrs,
    const float* __restrict__ chain_W,   // (3,32,32)
    const float* __restrict__ chain_b,   // (3,32)
    const float* __restrict__ lin_out_W, // (32,32)
    const float* __restrict__ lin_out_b, // (32,)
    const float* __restrict__ mlp_W1,    // (64,32)
    const float* __restrict__ mlp_b1,    // (64,)
    const float* __restrict__ prelu_a,   // (1,)
    const float* __restrict__ mlp_W2,    // (32,64)
    const float* __restrict__ mlp_b2,    // (32,)
    float* __restrict__ out)             // (B,T,N,F)
{
  // Pre-swizzled f16 B fragments (shared by all 8 waves):
  //  tiles 0..5  : chain_W[i], out-half p   (tile = 2*i+p)
  //  tiles 6..7  : lin_out_W, out-half p
  //  tiles 8..11 : mlp_W1, out-half p (o = 16p+n, o<64)
  //  tiles 12..15: mlp_W2, k-chunk q, out-half p (tile = 12+2q+p)
  __shared__ v16h fragB[16][32];
  // Per-wave activation transpose scratch: 16 rows x 64 cols f16
  __shared__ _Float16 sbuf[WAVES][16 * 64];

  const int tid  = threadIdx.x;
  const int wv   = tid >> 5;
  const int lane = tid & 31;
  const int hh   = lane >> 4;   // lane half
  const int nn   = lane & 15;   // column within tile (C layout) / A row

  // ---------------- build B fragments in LDS ----------------
  for (int fidx = tid; fidx < 16 * 32 * 8; fidx += 256) {
    const int t   = fidx >> 8;
    const int rem = fidx & 255;
    const int ln  = rem >> 3;
    const int j   = rem & 7;
    const int lh  = ln >> 4;
    const int lnn = ln & 15;
    // ISA 16-bit B layout (32x16): lane col n = l&15, half h = l>>4,
    // vgpr j halves hold rows K = 8h + 2j, +1 (j<4); K = 16 + 8h + 2(j-4), +1 (j>=4).
    const int k = (j < 4) ? (8 * lh + 2 * j) : (16 + 8 * lh + 2 * (j - 4));
    const float* src;
    if (t < 6) {
      const int i = t >> 1, p = t & 1;
      src = chain_W + i * F_ * F_ + (16 * p + lnn) * F_ + k;
    } else if (t < 8) {
      src = lin_out_W + (16 * (t - 6) + lnn) * F_ + k;
    } else if (t < 12) {
      src = mlp_W1 + (16 * (t - 8) + lnn) * F_ + k;
    } else {
      const int u = t - 12, q = u >> 1, p = u & 1;
      src = mlp_W2 + (16 * p + lnn) * (2 * F_) + 32 * q + k;
    }
    fragB[t][ln][2 * j]     = (_Float16)src[0];
    fragB[t][ln][2 * j + 1] = (_Float16)src[1];
  }
  __syncthreads();

  // ---------------- per-lane constants ----------------
  float al[LAYERS + 1], be[LAYERS + 1], th[LAYERS + 1];
#pragma unroll
  for (int i = 0; i <= LAYERS; ++i) {
    al[i] = clamp01(alphas[i]);
    be[i] = clamp01(betas[i]);
    th[i] = thrs[i];
  }
  const float pa = prelu_a[0];

  float bch[LAYERS][2], blo[2], bm1[4], bm2[2];
#pragma unroll
  for (int i = 0; i < LAYERS; ++i) {
    bch[i][0] = chain_b[i * F_ + nn];
    bch[i][1] = chain_b[i * F_ + 16 + nn];
  }
  blo[0] = lin_out_b[nn];
  blo[1] = lin_out_b[16 + nn];
#pragma unroll
  for (int p = 0; p < 4; ++p) bm1[p] = mlp_b1[16 * p + nn];
  bm2[0] = mlp_b2[nn];
  bm2[1] = mlp_b2[16 + nn];

  // ---------------- tile assignment ----------------
  const int tile = blockIdx.x * WAVES + wv;
  const int bb   = tile / TILES_PER_B;
  const int n0   = (tile % TILES_PER_B) * 16;
  _Float16* sb = sbuf[wv];

  // state (C/D layout: [layer][feature-half][row-in-lane]) -> registers
  float syn[LAYERS + 1][2][8];
  float mem[LAYERS + 1][2][8];
#pragma unroll
  for (int i = 0; i <= LAYERS; ++i)
#pragma unroll
    for (int f2 = 0; f2 < 2; ++f2)
#pragma unroll
      for (int r = 0; r < 8; ++r) { syn[i][f2][r] = 0.0f; mem[i][f2][r] = 0.0f; }

  // ---------------- time recurrence ----------------
  for (int t = 0; t < T_; ++t) {
    const size_t base = ((size_t)(bb * T_ + t) * N_ + n0) * F_;

    // load x tile in C layout
    float hcur[2][8];
#pragma unroll
    for (int f2 = 0; f2 < 2; ++f2)
#pragma unroll
      for (int r = 0; r < 8; ++r)
        hcur[f2][r] = x[base + (size_t)(8 * hh + r) * F_ + nn + 16 * f2];

    if (t + 1 < T_) {
      const size_t nbase = ((size_t)(bb * T_ + t + 1) * N_ + n0) * F_;
      __builtin_prefetch(x + nbase + lane * 16, 0, 3);  // global_prefetch_b8
    }

    // ---- chain layers: synaptic step + spike GEMM ----
#pragma unroll
    for (int i = 0; i < LAYERS; ++i) {
#pragma unroll
      for (int f2 = 0; f2 < 2; ++f2)
#pragma unroll
        for (int r = 0; r < 8; ++r) {
          float s = syn[i][f2][r];
          float m = mem[i][f2][r];
          const float reset = (m > th[i]) ? th[i] : 0.0f;  // reset * thr (detached)
          s = al[i] * s + hcur[f2][r];
          m = be[i] * m + s - reset;
          syn[i][f2][r] = s;
          mem[i][f2][r] = m;
          const float spk = (m > th[i]) ? 1.0f : 0.0f;
          sb[(8 * hh + r) * 64 + nn + 16 * f2] = (_Float16)spk;
        }
      const v16h a = loadA(sb, nn, hh, 0);
#pragma unroll
      for (int p = 0; p < 2; ++p) {
        const v8f c = wmma32_z(a, fragB[2 * i + p][lane]);  // inline-0 C
#pragma unroll
        for (int r = 0; r < 8; ++r) hcur[p][r] = c[r] + bch[i][p];  // bias post-add
      }
    }

    // ---- final synaptic layer + lin_out ----
    float tr[2][8];
    {
      const int i = LAYERS;
#pragma unroll
      for (int f2 = 0; f2 < 2; ++f2)
#pragma unroll
        for (int r = 0; r < 8; ++r) {
          float s = syn[i][f2][r];
          float m = mem[i][f2][r];
          const float reset = (m > th[i]) ? th[i] : 0.0f;
          s = al[i] * s + hcur[f2][r];
          m = be[i] * m + s - reset;
          syn[i][f2][r] = s;
          mem[i][f2][r] = m;
          const float spk = (m > th[i]) ? 1.0f : 0.0f;
          sb[(8 * hh + r) * 64 + nn + 16 * f2] = (_Float16)spk;
        }
      const v16h a = loadA(sb, nn, hh, 0);
#pragma unroll
      for (int p = 0; p < 2; ++p) {
        const v8f c = wmma32_z(a, fragB[6 + p][lane]);
#pragma unroll
        for (int r = 0; r < 8; ++r) tr[p][r] = c[r] + blo[p];
      }
    }

    // ---- MLP layer 1 (32 -> 64) + PReLU ----
#pragma unroll
    for (int f2 = 0; f2 < 2; ++f2)
#pragma unroll
      for (int r = 0; r < 8; ++r)
        sb[(8 * hh + r) * 64 + nn + 16 * f2] = (_Float16)tr[f2][r];
    const v16h aT = loadA(sb, nn, hh, 0);

    float h1[4][8];
#pragma unroll
    for (int p = 0; p < 4; ++p) {
      const v8f c = wmma32_z(aT, fragB[8 + p][lane]);
#pragma unroll
      for (int r = 0; r < 8; ++r) {
        const float v = c[r] + bm1[p];
        h1[p][r] = (v > 0.0f) ? v : pa * v;  // PReLU
      }
    }

    // ---- MLP layer 2 (64 -> 32) ----
#pragma unroll
    for (int p = 0; p < 4; ++p)
#pragma unroll
      for (int r = 0; r < 8; ++r)
        sb[(8 * hh + r) * 64 + 16 * p + nn] = (_Float16)h1[p][r];
    const v16h aq0 = loadA(sb, nn, hh, 0);
    const v16h aq1 = loadA(sb, nn, hh, 32);

#pragma unroll
    for (int p = 0; p < 2; ++p) {
      v8f c = wmma32_z(aq0, fragB[12 + p][lane]);   // K = 0..31, inline-0 C
      c = wmma32(aq1, fragB[14 + p][lane], c);      // K = 32..63 accumulates D->C
#pragma unroll
      for (int r = 0; r < 8; ++r)
        out[base + (size_t)(8 * hh + r) * F_ + 16 * p + nn] = c[r] + bm2[p];
    }
  }
}

extern "C" void kernel_launch(void* const* d_in, const int* in_sizes, int n_in,
                              void* d_out, int out_size, void* d_ws, size_t ws_size,
                              hipStream_t stream) {
  (void)in_sizes; (void)n_in; (void)out_size; (void)d_ws; (void)ws_size;
  const float* x         = (const float*)d_in[0];
  const float* alphas    = (const float*)d_in[1];
  const float* betas     = (const float*)d_in[2];
  const float* thrs      = (const float*)d_in[3];
  const float* chain_W   = (const float*)d_in[4];
  const float* chain_b   = (const float*)d_in[5];
  const float* lin_out_W = (const float*)d_in[6];
  const float* lin_out_b = (const float*)d_in[7];
  const float* mlp_W1    = (const float*)d_in[8];
  const float* mlp_b1    = (const float*)d_in[9];
  const float* prelu_a   = (const float*)d_in[10];
  const float* mlp_W2    = (const float*)d_in[11];
  const float* mlp_b2    = (const float*)d_in[12];
  float* out             = (float*)d_out;

  const int total_tiles = (B_ * N_) / 16;        // 1024
  const int blocks      = total_tiles / WAVES;   // 128 blocks of 8 waves
  snn_chain_kernel<<<blocks, 256, 0, stream>>>(
      x, alphas, betas, thrs, chain_W, chain_b, lin_out_W, lin_out_b,
      mlp_W1, mlp_b1, prelu_a, mlp_W2, mlp_b2, out);
}
